// RegistrationDecoder_71253507440717
// MI455X (gfx1250) — matrix operand order repaired and checked by reading, not compiled, not run on hardware
//
#include <hip/hip_runtime.h>
#include <math.h>

// B=65536, C=16, L=32, E=32.  Output: [B,3] f32.
// Pass1: S[c,d] = sum_b exp(qhat.khatT)   (block partials -> reduce kernel)
// Pass2: full pipeline per batch, WMMA everywhere; MLP split across 3 waves.

typedef __attribute__((ext_vector_type(16))) _Float16 v16h;
typedef __attribute__((ext_vector_type(8)))  float    v8f;

#define WMMA_F16(a, b, c) \
  __builtin_amdgcn_wmma_f32_16x16x32_f16(false, (a), false, (b), (short)0, (c), false, false)

// ---- layout helpers -------------------------------------------------------
// 16-bit A matrix 16x32 (M x K): lane m = lane&15; halves h: K=(h<8?h:h+8)+(lane>=16?8:0)
__device__ __forceinline__ v16h load_A_f16(const float* __restrict__ src /*[16][32] row-major*/) {
  const int lane = threadIdx.x & 31;
  const int m = lane & 15;
  const int hi = lane >> 4;
  v16h a;
#pragma unroll
  for (int h = 0; h < 16; ++h) {
    int k = (h < 8 ? h : h + 8) + hi * 8;
    a[h] = (_Float16)src[m * 32 + k];
  }
  return a;
}

// 16-bit B matrix 32x16 (K x N) from W [E=32][L=32] as B[k=l][n=e]=W[e][l]; etile picks e range
__device__ __forceinline__ v16h load_WT_B_f16(const float* __restrict__ W, int etile) {
  const int lane = threadIdx.x & 31;
  const int n = (lane & 15) + etile * 16;
  const int hi = lane >> 4;
  v16h b;
#pragma unroll
  for (int h = 0; h < 16; ++h) {
    int k = (h < 8 ? h : h + 8) + hi * 8;
    b[h] = (_Float16)W[n * 32 + k];
  }
  return b;
}

// C/D layout: lane n = lane&15, reg r holds row M = r + (lane>=16 ? 8 : 0).
// Normalize rows of a 16x32 matrix held as two 16x16 f32 tiles.
__device__ __forceinline__ void row_normalize(v8f& t0, v8f& t1) {
#pragma unroll
  for (int r = 0; r < 8; ++r) {
    float s = t0[r] * t0[r] + t1[r] * t1[r];
    s += __shfl_xor(s, 1);
    s += __shfl_xor(s, 2);
    s += __shfl_xor(s, 4);
    s += __shfl_xor(s, 8);   // full 16-lane (one-row) reduction within each half-wave
    float inv = 1.0f / fmaxf(sqrtf(s), 1e-12f);
    t0[r] *= inv;
    t1[r] *= inv;
  }
}

// prefetch a 2KB embed tile, one 64B line per lane (lowers to global_prefetch_b8)
__device__ __forceinline__ void prefetch_tile(const float* p) {
  const int lane = threadIdx.x & 31;
  __builtin_prefetch(p + lane * 16, 0, 1);
}

// ---- pass 1: softmax denominators ----------------------------------------
#define P1_BLOCKS 1024   // 1024 blocks * 8 waves * 8 iters = 65536 batches

__global__ __launch_bounds__(256) void pass1_sums(
    const float* __restrict__ e1, const float* __restrict__ e2,
    const float* __restrict__ qW, const float* __restrict__ qb,
    const float* __restrict__ kW, const float* __restrict__ kb,
    float* __restrict__ partial)
{
  __shared__ float s_q[8][512];
  __shared__ float s_k[8][512];
  __shared__ float s_red[8][256];

  const int lane = threadIdx.x & 31;
  const int wid  = threadIdx.x >> 5;
  const int n15  = lane & 15;
  const int hi   = lane >> 4;

  v16h qB0 = load_WT_B_f16(qW, 0), qB1 = load_WT_B_f16(qW, 1);
  v16h kB0 = load_WT_B_f16(kW, 0), kB1 = load_WT_B_f16(kW, 1);
  const float qbias0 = qb[n15], qbias1 = qb[16 + n15];
  const float kbias0 = kb[n15], kbias1 = kb[16 + n15];

  float* myq = s_q[wid];
  float* myk = s_k[wid];
  v8f zero = {};
  v8f acc  = {};

  const int gw = blockIdx.x * 8 + wid;
  for (int it = 0; it < 8; ++it) {
    const int b = gw + it * (P1_BLOCKS * 8);
    if (it + 1 < 8) {   // hide HBM latency of the next tile behind this WMMA chain
      const int bn = gw + (it + 1) * (P1_BLOCKS * 8);
      prefetch_tile(e1 + (size_t)bn * 512);
      prefetch_tile(e2 + (size_t)bn * 512);
    }
    v16h a1 = load_A_f16(e1 + (size_t)b * 512);
    v16h a2 = load_A_f16(e2 + (size_t)b * 512);

    v8f q0 = WMMA_F16(a1, qB0, zero), q0b = WMMA_F16(a2, qB0, zero);
    v8f q1 = WMMA_F16(a1, qB1, zero), q1b = WMMA_F16(a2, qB1, zero);
    v8f k0 = WMMA_F16(a1, kB0, zero), k0b = WMMA_F16(a2, kB0, zero);
    v8f k1 = WMMA_F16(a1, kB1, zero), k1b = WMMA_F16(a2, kB1, zero);
#pragma unroll
    for (int r = 0; r < 8; ++r) {
      q0[r] = (q0[r] + qbias0) * (q0b[r] + qbias0);
      q1[r] = (q1[r] + qbias1) * (q1b[r] + qbias1);
      k0[r] = (k0[r] + kbias0) * (k0b[r] + kbias0);
      k1[r] = (k1[r] + kbias1) * (k1b[r] + kbias1);
    }
    row_normalize(q0, q1);
    row_normalize(k0, k1);

    // stage qhat/khat to LDS as [c][e] f32
#pragma unroll
    for (int r = 0; r < 8; ++r) {
      int c = r + hi * 8;
      myq[c * 32 + n15]      = q0[r];
      myq[c * 32 + 16 + n15] = q1[r];
      myk[c * 32 + n15]      = k0[r];
      myk[c * 32 + 16 + n15] = k1[r];
    }
    // attn = qhat @ khat^T : A rows = c, K = e;  B[k=e][n=d] = khat[d][e]
    v16h aq, bk;
#pragma unroll
    for (int h = 0; h < 16; ++h) {
      int ka = (h < 8 ? h : h + 8) + hi * 8;
      aq[h] = (_Float16)myq[n15 * 32 + ka];
      int kb2 = (h < 8 ? h : h + 8) + hi * 8;
      bk[h] = (_Float16)myk[n15 * 32 + kb2];
    }
    v8f attn = WMMA_F16(aq, bk, zero);
#pragma unroll
    for (int r = 0; r < 8; ++r) acc[r] += __expf(attn[r]);   // |attn|<=1 -> no max shift needed
  }

  // deterministic block reduction of the 8 waves
#pragma unroll
  for (int r = 0; r < 8; ++r) s_red[wid][lane * 8 + r] = acc[r];
  __syncthreads();
  {
    const int t = threadIdx.x;
    const int src_lane = t >> 3;
    const int r = t & 7;
    float s = 0.f;
#pragma unroll
    for (int w = 0; w < 8; ++w) s += s_red[w][src_lane * 8 + r];
    const int c = r + ((src_lane >> 4) << 3);
    const int d = src_lane & 15;
    partial[(size_t)blockIdx.x * 256 + c * 16 + d] = s;
  }
}

__global__ __launch_bounds__(256) void reduce_sums(const float* __restrict__ partial,
                                                   float* __restrict__ S)
{
  const int t = threadIdx.x;   // 256 positions
  float s = 0.f;
  for (int i = 0; i < P1_BLOCKS; ++i) s += partial[(size_t)i * 256 + t];
  S[t] = s;
}

// ---- pass 2: full pipeline -----------------------------------------------
#define P2_BLOCKS 2048   // 2048 blocks * 8 waves * 4 iters = 65536 batches

__global__ __launch_bounds__(256) void pass2_main(
    const float* __restrict__ e1, const float* __restrict__ e2,
    const float* __restrict__ qW, const float* __restrict__ qb,
    const float* __restrict__ kW, const float* __restrict__ kb,
    const float* __restrict__ vW, const float* __restrict__ vb,
    const float* __restrict__ p1W, const float* __restrict__ p1b,
    const float* __restrict__ p2W, const float* __restrict__ p2b,
    const float* __restrict__ S,  float* __restrict__ out)
{
  __shared__ float s_q[8][512];    // qhat, later value [c][e]
  __shared__ float s_k[8][512];    // khat, later attnW [c][d] (first 256)
  __shared__ float s_out[8][512];  // per-batch out, flat d*32+e
  __shared__ float s_h[16][48];

  const int lane = threadIdx.x & 31;
  const int wid  = threadIdx.x >> 5;
  const int n15  = lane & 15;
  const int hi   = lane >> 4;

  v16h qB0 = load_WT_B_f16(qW, 0), qB1 = load_WT_B_f16(qW, 1);
  v16h kB0 = load_WT_B_f16(kW, 0), kB1 = load_WT_B_f16(kW, 1);
  v16h vB0 = load_WT_B_f16(vW, 0), vB1 = load_WT_B_f16(vW, 1);
  const float qbias0 = qb[n15], qbias1 = qb[16 + n15];
  const float kbias0 = kb[n15], kbias1 = kb[16 + n15];
  const float vbias0 = vb[n15], vbias1 = vb[16 + n15];

  v8f sinv;  // reciprocal softmax denominators, in attn's C/D layout
#pragma unroll
  for (int r = 0; r < 8; ++r) sinv[r] = 1.0f / S[(r + hi * 8) * 16 + n15];

  float* myq = s_q[wid];
  float* myk = s_k[wid];
  v8f zero = {};

  const int gw = blockIdx.x * 8 + wid;
  for (int it = 0; it < 4; ++it) {
    const int b = gw + it * (P2_BLOCKS * 8);
    if (it + 1 < 4) {
      const int bn = gw + (it + 1) * (P2_BLOCKS * 8);
      prefetch_tile(e1 + (size_t)bn * 512);
      prefetch_tile(e2 + (size_t)bn * 512);
    }
    v16h a1 = load_A_f16(e1 + (size_t)b * 512);
    v16h a2 = load_A_f16(e2 + (size_t)b * 512);

    v8f q0 = WMMA_F16(a1, qB0, zero), q0b = WMMA_F16(a2, qB0, zero);
    v8f q1 = WMMA_F16(a1, qB1, zero), q1b = WMMA_F16(a2, qB1, zero);
    v8f k0 = WMMA_F16(a1, kB0, zero), k0b = WMMA_F16(a2, kB0, zero);
    v8f k1 = WMMA_F16(a1, kB1, zero), k1b = WMMA_F16(a2, kB1, zero);
    v8f v0 = WMMA_F16(a1, vB0, zero), v0b = WMMA_F16(a2, vB0, zero);
    v8f v1 = WMMA_F16(a1, vB1, zero), v1b = WMMA_F16(a2, vB1, zero);
#pragma unroll
    for (int r = 0; r < 8; ++r) {
      q0[r] = (q0[r] + qbias0) * (q0b[r] + qbias0);
      q1[r] = (q1[r] + qbias1) * (q1b[r] + qbias1);
      k0[r] = (k0[r] + kbias0) * (k0b[r] + kbias0);
      k1[r] = (k1[r] + kbias1) * (k1b[r] + kbias1);
      v0[r] = (v0[r] + vbias0) * (v0b[r] + vbias0);
      v1[r] = (v1[r] + vbias1) * (v1b[r] + vbias1);
    }
    row_normalize(q0, q1);
    row_normalize(k0, k1);

#pragma unroll
    for (int r = 0; r < 8; ++r) {
      int c = r + hi * 8;
      myq[c * 32 + n15]      = q0[r];
      myq[c * 32 + 16 + n15] = q1[r];
      myk[c * 32 + n15]      = k0[r];
      myk[c * 32 + 16 + n15] = k1[r];
    }
    v16h aq, bk;
#pragma unroll
    for (int h = 0; h < 16; ++h) {
      int ka = (h < 8 ? h : h + 8) + hi * 8;
      aq[h] = (_Float16)myq[n15 * 32 + ka];
      int kb2 = (h < 8 ? h : h + 8) + hi * 8;
      bk[h] = (_Float16)myk[n15 * 32 + kb2];
    }
    v8f attn = WMMA_F16(aq, bk, zero);
#pragma unroll
    for (int r = 0; r < 8; ++r) attn[r] = __expf(attn[r]) * sinv[r];  // attnW

    // stage attnW -> myk[c*16+d], value -> myq[c*32+e]  (LDS ops stay in order per wave)
#pragma unroll
    for (int r = 0; r < 8; ++r) {
      int c = r + hi * 8;
      myk[c * 16 + n15]      = attn[r];
      myq[c * 32 + n15]      = v0[r];
      myq[c * 32 + 16 + n15] = v1[r];
    }
    // out[d][e] = sum_c attnW[c][d] * value[c][e]; K=16 zero-padded to 32
    v16h aT, bV0, bV1;
#pragma unroll
    for (int h = 0; h < 16; ++h) {
      int ka = (h < 8 ? h : h + 8) + hi * 8;
      aT[h] = (ka < 16) ? (_Float16)myk[ka * 16 + n15] : (_Float16)0.0f;
      int kb2 = (h < 8 ? h : h + 8) + hi * 8;
      bV0[h] = (kb2 < 16) ? (_Float16)myq[kb2 * 32 + n15]      : (_Float16)0.0f;
      bV1[h] = (kb2 < 16) ? (_Float16)myq[kb2 * 32 + 16 + n15] : (_Float16)0.0f;
    }
    v8f o0 = WMMA_F16(aT, bV0, zero);
    v8f o1 = WMMA_F16(aT, bV1, zero);
#pragma unroll
    for (int r = 0; r < 8; ++r) {
      int d = r + hi * 8;
      s_out[wid][d * 32 + n15]      = o0[r];
      s_out[wid][d * 32 + 16 + n15] = o1[r];
    }
    __syncthreads();

    // MLP for the block's 8 batches: M=16 (rows 8..15 zero) x K=512 x N=48.
    // The three N-tiles are independent -> waves 0..2 each own one (wave-uniform branch,
    // EXEC stays all-ones for WMMA).
    if (wid < 3) {
      const int nt = wid;
      const int j = nt * 16 + n15;
      v8f hacc = zero;
      for (int kc = 0; kc < 16; ++kc) {
        v16h aH, bW;
#pragma unroll
        for (int h = 0; h < 16; ++h) {
          int kl = (h < 8 ? h : h + 8) + hi * 8;
          aH[h] = (n15 < 8) ? (_Float16)s_out[n15][kc * 32 + kl] : (_Float16)0.0f;
          int i = kc * 32 + (h < 8 ? h : h + 8) + hi * 8;
          bW[h] = (_Float16)p1W[(size_t)j * 512 + i];
        }
        hacc = WMMA_F16(aH, bW, hacc);
      }
      const float pb = p1b[j];
#pragma unroll
      for (int r = 0; r < 8; ++r) {
        int m = r + hi * 8;
        s_h[m][j] = fmaxf(hacc[r] + pb, 0.f);
      }
    }
    __syncthreads();

    if (wid == 0 && lane < 8) {   // 3x48 projection + L2-normalize, one lane per batch
      const int bb = lane;
      float y0 = p2b[0], y1 = p2b[1], y2 = p2b[2];
      for (int jj = 0; jj < 48; ++jj) {
        float hv = s_h[bb][jj];
        y0 += hv * p2W[jj];
        y1 += hv * p2W[48 + jj];
        y2 += hv * p2W[96 + jj];
      }
      float inv = 1.0f / fmaxf(sqrtf(y0 * y0 + y1 * y1 + y2 * y2), 1e-12f);
      size_t bg = (size_t)(blockIdx.x * 8 + bb) + (size_t)it * (P2_BLOCKS * 8);
      out[bg * 3 + 0] = y0 * inv;
      out[bg * 3 + 1] = y1 * inv;
      out[bg * 3 + 2] = y2 * inv;
    }
    __syncthreads();
  }
}

// ---- launch ---------------------------------------------------------------
extern "C" void kernel_launch(void* const* d_in, const int* in_sizes, int n_in,
                              void* d_out, int out_size, void* d_ws, size_t ws_size,
                              hipStream_t stream) {
  const float* embed1 = (const float*)d_in[0];
  const float* embed2 = (const float*)d_in[1];
  const float* qW = (const float*)d_in[2];
  const float* qb = (const float*)d_in[3];
  const float* kW = (const float*)d_in[4];
  const float* kb = (const float*)d_in[5];
  const float* vW = (const float*)d_in[6];
  const float* vb = (const float*)d_in[7];
  const float* p1W = (const float*)d_in[8];
  const float* p1b = (const float*)d_in[9];
  const float* p2W = (const float*)d_in[10];
  const float* p2b = (const float*)d_in[11];
  float* outp = (float*)d_out;

  float* partial = (float*)d_ws;                      // P1_BLOCKS*256 floats (1 MB)
  float* S = partial + (size_t)P1_BLOCKS * 256;       // 256 floats

  pass1_sums<<<P1_BLOCKS, 256, 0, stream>>>(embed1, embed2, qW, qb, kW, kb, partial);
  reduce_sums<<<1, 256, 0, stream>>>(partial, S);
  pass2_main<<<P2_BLOCKS, 256, 0, stream>>>(embed1, embed2, qW, qb, kW, kb, vW, vb,
                                            p1W, p1b, p2W, p2b, S, outp);
}
